// MultiHeadAttention_12807592477064
// MI455X (gfx1250) — matrix operand order repaired and checked by reading, not compiled, not run on hardware
//
#include <hip/hip_runtime.h>
#include <hip/hip_bf16.h>

// ---------------------------------------------------------------------------
// MI455X (gfx1250) multi-head attention
//   bf16 WMMA + TDM (tensor_load_to_lds) double-buffered LDS staging
//   batched ds_load fragments -> back-to-back WMMA chains
// ---------------------------------------------------------------------------

#define S_LEN 4096
#define D_MODEL 1024
#define N_HEADS 16
#define D_K 64

typedef __bf16 bf16_t;
typedef __attribute__((ext_vector_type(16))) __bf16 v16bf;
typedef __attribute__((ext_vector_type(8)))  float  v8f;
typedef __attribute__((ext_vector_type(4)))  unsigned uint32x4;
typedef __attribute__((ext_vector_type(8)))  int      int32x8;
typedef __attribute__((ext_vector_type(4)))  int      int32x4;

union AB32 { v16bf v; uint4 q[2]; };   // 32-byte WMMA A/B fragment

__device__ __forceinline__ bf16_t f2bf(float f) {
  unsigned u = __builtin_bit_cast(unsigned, f);
  u += 0x7FFFu + ((u >> 16) & 1u);               // round-to-nearest-even
  unsigned short s = (unsigned short)(u >> 16);
  return __builtin_bit_cast(bf16_t, s);
}

// generic pointer (to a __shared__ object) -> LDS byte offset
typedef __attribute__((address_space(3))) unsigned short lds_us_t;
__device__ __forceinline__ unsigned lds_off_of(unsigned short* p) {
  return (unsigned)(unsigned long long)(lds_us_t*)p;
}

// ---- Tensor Data Mover: 2D tile (bf16) global -> LDS, optional LDS padding -
// tile_d0 = elements per row, tile_d1 = rows, stride_elems = row stride.
// pad: insert pad_amount(=code+1) DWORDs after every 2^(code+1) DWORDs stored.
__device__ __forceinline__ void tdm_load_2d(unsigned lds_byte_off, const void* gsrc,
                                            unsigned tile_d0, unsigned tile_d1,
                                            unsigned stride_elems,
                                            unsigned pad_interval_code,
                                            unsigned pad_amount_code,
                                            bool pad_en)
{
  unsigned long long ga = (unsigned long long)gsrc;
  uint32x4 g0;
  g0[0] = 1u;                                        // count=1, no gather
  g0[1] = lds_byte_off;                              // lds_addr
  g0[2] = (unsigned)ga;                              // global_addr[31:0]
  g0[3] = ((unsigned)(ga >> 32) & 0x01FFFFFFu) | (2u << 30);  // addr[56:32], type=2

  unsigned w0 = (1u << 16);                          // data_size = 1 (2 bytes)
  if (pad_en) w0 |= (1u << 20) | (pad_interval_code << 22) | (pad_amount_code << 25);
  int32x8 g1;
  g1[0] = (int)w0;                                   // mask=0, ds=2B, pad cfg
  g1[1] = (int)((tile_d0 & 0xFFFFu) << 16);          // tensor_dim0[15:0] @48
  g1[2] = (int)((tile_d0 >> 16) | ((tile_d1 & 0xFFFFu) << 16)); // td0 hi, td1 lo
  g1[3] = (int)((tile_d1 >> 16) | (tile_d0 << 16));  // td1 hi, tile_dim0 @112
  g1[4] = (int)(tile_d1 & 0xFFFFu);                  // tile_dim1; tile_dim2=0
  g1[5] = (int)stride_elems;                         // tensor_dim0_stride lo32
  g1[6] = 0;                                         // stride hi, dim1_stride=0
  g1[7] = 0;

  int32x4 gz4; gz4[0] = 0; gz4[1] = 0; gz4[2] = 0; gz4[3] = 0;
  int32x8 gz8;
  gz8[0] = 0; gz8[1] = 0; gz8[2] = 0; gz8[3] = 0;
  gz8[4] = 0; gz8[5] = 0; gz8[6] = 0; gz8[7] = 0;
  // 6-arg form (clang-23 / therock headers): g0, g1, g2, g3, extra, cpol
  __builtin_amdgcn_tensor_load_to_lds(g0, g1, gz4, gz4, gz8, 0);
}

// ---------------- conversion kernels ----------------

__global__ __launch_bounds__(256) void cvt_f32_bf16(const float* __restrict__ in,
                                                    bf16_t* __restrict__ out, int n) {
  int i = blockIdx.x * blockDim.x + threadIdx.x;
  if (i < n) out[i] = f2bf(in[i]);
}

// out[n*K + k] = in[k*N + n]   (W [K,N] f32 -> Wt [N,K] bf16)
__global__ __launch_bounds__(256) void cvt_transpose_bf16(const float* __restrict__ in,
                                                          bf16_t* __restrict__ out,
                                                          int K, int N) {
  int i = blockIdx.x * blockDim.x + threadIdx.x;
  if (i < K * N) {
    int nn = i / K;
    int k  = i % K;
    out[i] = f2bf(in[(size_t)k * N + nn]);
  }
}

// ---------------- WMMA GEMM: C = A[M,K] @ Wt[N,K]^T + bias ------------------
// Block: 8 waves, tile 128(M) x 64(N). B K-chunks (64x64) TDM-staged in LDS,
// double buffered, shared by all waves. A per-wave from global.
// mode 0: out bf16 [M,N] | mode 1: out bf16 [N,M] (packed) | mode 2: out f32 [M,N]
__global__ __launch_bounds__(256) void gemm_bf16_wmma(
    const bf16_t* __restrict__ A, const bf16_t* __restrict__ Wt,
    const float* __restrict__ bias, void* __restrict__ outp,
    int M, int N, int K, int mode)
{
  __shared__ __align__(16) unsigned short Bld[2][64 * 72];   // pitch 72 elems
  const int lane = threadIdx.x & 31;
  const int w    = threadIdx.x >> 5;
  const int nGroups = N >> 6;
  const int mBlk   = blockIdx.x / nGroups;
  const int nGroup = blockIdx.x % nGroups;
  const int mBase = mBlk * 128 + w * 16;
  const int nBase = nGroup << 6;
  const int half = lane >> 4;
  const int lidx = lane & 15;
  const int ksteps = K >> 6;                 // 64-wide K chunks

  if (w == 0)   // prologue: stage chunk 0
    tdm_load_2d(lds_off_of(&Bld[0][0]), Wt + (size_t)nBase * K,
                64, 64, (unsigned)K, 4, 3, true);

  v8f acc[4] = {};
  const bf16_t* arow = A + (size_t)(mBase + lidx) * K;

  for (int ks = 0; ks < ksteps; ++ks) {
    const int cur = ks & 1;
    // A fragments for both 32-wide sub-chunks: issue early (global, no barrier dep)
    AB32 a0, a1;
    const bf16_t* ap = arow + (ks << 6);
    a0.q[0] = *(const uint4*)(ap + 8 * half);
    a0.q[1] = *(const uint4*)(ap + 16 + 8 * half);
    a1.q[0] = *(const uint4*)(ap + 32 + 8 * half);
    a1.q[1] = *(const uint4*)(ap + 48 + 8 * half);

    if (w == 0) {
      if (ks + 1 < ksteps) {
        tdm_load_2d(lds_off_of(&Bld[cur ^ 1][0]),
                    Wt + (size_t)nBase * K + (size_t)(ks + 1) * 64,
                    64, 64, (unsigned)K, 4, 3, true);
        __builtin_amdgcn_s_wait_tensorcnt(1);   // chunk ks landed
      } else {
        __builtin_amdgcn_s_wait_tensorcnt(0);
      }
    }
    __syncthreads();

    // Batch ALL B-fragment LDS loads first, then chain the 8 WMMAs.
    AB32 b[8];
#pragma unroll
    for (int c = 0; c < 2; ++c) {
#pragma unroll
      for (int t = 0; t < 4; ++t) {
        unsigned short* bp = &Bld[cur][((t << 4) + lidx) * 72 + (c << 5) + 8 * half];
        b[c * 4 + t].q[0] = *(const uint4*)(bp);
        b[c * 4 + t].q[1] = *(const uint4*)(bp + 16);
      }
    }
#pragma unroll
    for (int c = 0; c < 2; ++c) {
      const AB32& a = c ? a1 : a0;
#pragma unroll
      for (int t = 0; t < 4; ++t)
        acc[t] = __builtin_amdgcn_wmma_f32_16x16x32_bf16(
                     false, a.v, false, b[c * 4 + t].v, (short)0, acc[t], false, false);
    }
    __syncthreads();   // all waves done reading before buffer is overwritten
  }

#pragma unroll
  for (int t = 0; t < 4; ++t) {
    const int nG = nBase + (t << 4) + lidx;
    const float bv = bias[nG];
    if (mode == 0) {
      bf16_t* out = (bf16_t*)outp;
#pragma unroll
      for (int r = 0; r < 8; ++r)
        out[(size_t)(mBase + 8 * half + r) * N + nG] = f2bf(acc[t][r] + bv);
    } else if (mode == 1) {
      bf16_t* out = (bf16_t*)outp;
      unsigned pw[4];
#pragma unroll
      for (int i = 0; i < 4; ++i) {
        unsigned lo = (unsigned)__builtin_bit_cast(unsigned short, f2bf(acc[t][2 * i]     + bv));
        unsigned hi = (unsigned)__builtin_bit_cast(unsigned short, f2bf(acc[t][2 * i + 1] + bv));
        pw[i] = lo | (hi << 16);
      }
      uint4 pk; pk.x = pw[0]; pk.y = pw[1]; pk.z = pw[2]; pk.w = pw[3];
      *(uint4*)(out + (size_t)nG * M + mBase + 8 * half) = pk;
    } else {
      float* out = (float*)outp;
#pragma unroll
      for (int r = 0; r < 8; ++r)
        out[(size_t)(mBase + 8 * half + r) * N + nG] = acc[t][r] + bv;
    }
  }
}

// ---------------- flash attention -------------------------------------------
// Block: one head, 128 queries (8 waves x 16). K/V^T kv-tiles (32 keys) are
// TDM-staged into LDS, double buffered, shared by all 8 waves.
__global__ __launch_bounds__(256) void attn_fwd_wmma(
    const bf16_t* __restrict__ Qb, const bf16_t* __restrict__ Kb,
    const bf16_t* __restrict__ Vt, bf16_t* __restrict__ O)
{
  __shared__ __align__(16) unsigned short Kld[2][32 * 72];  // 32 kv x 64 d, pitch 72
  __shared__ __align__(16) unsigned short Vld[2][64 * 40];  // 64 d x 32 kv, pitch 40
  __shared__ __align__(16) unsigned short Pbuf[8][16][32];  // per-wave P re-layout
  const int lane = threadIdx.x & 31;
  const int w    = threadIdx.x >> 5;
  const int head = blockIdx.x >> 5;       // 16 heads x 32 q-blocks
  const int qBlk = blockIdx.x & 31;
  const int qBase = qBlk * 128 + w * 16;
  const int hd    = head << 6;
  const int half = lane >> 4;
  const int n    = lane & 15;

  // Q fragments for the two 32-wide d_k steps (global, loaded once)
  AB32 qf[2];
  const bf16_t* qrow = Qb + (size_t)(qBase + n) * D_MODEL + hd;
#pragma unroll
  for (int s = 0; s < 2; ++s) {
    qf[s].q[0] = *(const uint4*)(qrow + 32 * s + 8 * half);
    qf[s].q[1] = *(const uint4*)(qrow + 32 * s + 16 + 8 * half);
  }

  const int nkv = qBlk * 4 + 4;           // causal: kv tiles of 32 for whole block

  if (w == 0) {                           // prologue: stage tile 0
    tdm_load_2d(lds_off_of(&Kld[0][0]), Kb + hd, 64, 32, D_MODEL, 4, 3, true);
    tdm_load_2d(lds_off_of(&Vld[0][0]), Vt + (size_t)hd * S_LEN, 32, 64, S_LEN, 3, 3, true);
  }

  v8f acc[4] = {};
  float mrow[8], lrow[8];
#pragma unroll
  for (int r = 0; r < 8; ++r) { mrow[r] = -3.0e38f; lrow[r] = 0.0f; }

  for (int kt = 0; kt < nkv; ++kt) {
    const int kvBase = kt << 5;
    const int cur = kt & 1;

    if (w == 0) {
      if (kt + 1 < nkv) {
        const int kvN = (kt + 1) << 5;
        tdm_load_2d(lds_off_of(&Kld[cur ^ 1][0]),
                    Kb + (size_t)kvN * D_MODEL + hd, 64, 32, D_MODEL, 4, 3, true);
        tdm_load_2d(lds_off_of(&Vld[cur ^ 1][0]),
                    Vt + (size_t)hd * S_LEN + kvN, 32, 64, S_LEN, 3, 3, true);
        __builtin_amdgcn_s_wait_tensorcnt(2);   // tile kt's two DMAs landed
      } else {
        __builtin_amdgcn_s_wait_tensorcnt(0);
      }
    }
    __syncthreads();

    // ---- scores: S = Q @ K^T from LDS (batched loads, then WMMA chain) ----
    AB32 kbf[4];
#pragma unroll
    for (int sub = 0; sub < 2; ++sub) {
#pragma unroll
      for (int s = 0; s < 2; ++s) {
        unsigned short* kp = &Kld[cur][((sub << 4) + n) * 72 + 32 * s + 8 * half];
        kbf[sub * 2 + s].q[0] = *(const uint4*)(kp);
        kbf[sub * 2 + s].q[1] = *(const uint4*)(kp + 16);
      }
    }
    v8f sc[2] = {};
#pragma unroll
    for (int sub = 0; sub < 2; ++sub) {
#pragma unroll
      for (int s = 0; s < 2; ++s)
        sc[sub] = __builtin_amdgcn_wmma_f32_16x16x32_bf16(
                      false, qf[s].v, false, kbf[sub * 2 + s].v, (short)0, sc[sub], false, false);
    }

    // ---- online softmax ----
    float p0[8], p1[8];
#pragma unroll
    for (int r = 0; r < 8; ++r) {
      const int qG = qBase + 8 * half + r;
      float a0 = sc[0][r] * 0.125f;             // 1/sqrt(64)
      float a1 = sc[1][r] * 0.125f;
      if (kvBase + n > qG)       a0 = -3.0e38f; // causal mask
      if (kvBase + 16 + n > qG)  a1 = -3.0e38f;
      float rm = fmaxf(a0, a1);
      rm = fmaxf(rm, __shfl_xor(rm, 1, 32));
      rm = fmaxf(rm, __shfl_xor(rm, 2, 32));
      rm = fmaxf(rm, __shfl_xor(rm, 4, 32));
      rm = fmaxf(rm, __shfl_xor(rm, 8, 32));
      const float mnew = fmaxf(mrow[r], rm);
      const float corr = __expf(mrow[r] - mnew);
      mrow[r] = mnew;
      const float e0 = __expf(a0 - mnew);
      const float e1 = __expf(a1 - mnew);
      float rs = e0 + e1;
      rs += __shfl_xor(rs, 1, 32);
      rs += __shfl_xor(rs, 2, 32);
      rs += __shfl_xor(rs, 4, 32);
      rs += __shfl_xor(rs, 8, 32);
      lrow[r] = lrow[r] * corr + rs;
#pragma unroll
      for (int t = 0; t < 4; ++t) acc[t][r] *= corr;
      p0[r] = e0; p1[r] = e1;
    }

    // ---- re-layout P (C-frag -> A-frag) via LDS; V frags batched alongside --
#pragma unroll
    for (int r = 0; r < 8; ++r) {
      Pbuf[w][8 * half + r][n]      = __builtin_bit_cast(unsigned short, f2bf(p0[r]));
      Pbuf[w][8 * half + r][16 + n] = __builtin_bit_cast(unsigned short, f2bf(p1[r]));
    }
    AB32 vbf[4];
#pragma unroll
    for (int t = 0; t < 4; ++t) {
      unsigned short* vp = &Vld[cur][((t << 4) + n) * 40 + 8 * half];
      vbf[t].q[0] = *(const uint4*)(vp);
      vbf[t].q[1] = *(const uint4*)(vp + 16);
    }
    asm volatile("s_wait_dscnt 0" ::: "memory");
    AB32 apf;
    apf.q[0] = *(const uint4*)(&Pbuf[w][n][8 * half]);
    apf.q[1] = *(const uint4*)(&Pbuf[w][n][16 + 8 * half]);

    // ---- O += P @ V (back-to-back WMMAs) ----
#pragma unroll
    for (int t = 0; t < 4; ++t)
      acc[t] = __builtin_amdgcn_wmma_f32_16x16x32_bf16(
                   false, apf.v, false, vbf[t].v, (short)0, acc[t], false, false);
    __syncthreads();   // done reading tile kt before its buffer is re-filled
  }

  // ---- epilogue: normalize and store bf16 attention output ----
#pragma unroll
  for (int t = 0; t < 4; ++t) {
#pragma unroll
    for (int r = 0; r < 8; ++r) {
      float o = acc[t][r] / lrow[r];
      O[(size_t)(qBase + 8 * half + r) * D_MODEL + hd + (t << 4) + n] = f2bf(o);
    }
  }
}

// ---------------------------------------------------------------------------

extern "C" void kernel_launch(void* const* d_in, const int* in_sizes, int n_in,
                              void* d_out, int out_size, void* d_ws, size_t ws_size,
                              hipStream_t stream) {
  (void)in_sizes; (void)n_in; (void)out_size; (void)ws_size;
  const float* x    = (const float*)d_in[0];
  // d_in[1] = causal tril mask (int32) — applied analytically in attn kernel
  const float* Wq_w = (const float*)d_in[2];
  const float* Wq_b = (const float*)d_in[3];
  const float* Wk_w = (const float*)d_in[4];
  const float* Wk_b = (const float*)d_in[5];
  const float* Wv_w = (const float*)d_in[6];
  const float* Wv_b = (const float*)d_in[7];
  const float* Wo_w = (const float*)d_in[8];
  const float* Wo_b = (const float*)d_in[9];

  char* ws = (char*)d_ws;
  const size_t MB = 1024ull * 1024ull;
  bf16_t* xb  = (bf16_t*)(ws + 0);        //  8 MB  x   bf16 [S,D]
  bf16_t* wtq = (bf16_t*)(ws + 8  * MB);  //  2 MB  Wq^T bf16 [D,D]
  bf16_t* wtk = (bf16_t*)(ws + 10 * MB);
  bf16_t* wtv = (bf16_t*)(ws + 12 * MB);
  bf16_t* wto = (bf16_t*)(ws + 14 * MB);
  bf16_t* Qb  = (bf16_t*)(ws + 16 * MB);  //  8 MB  Q bf16 [S,D]
  bf16_t* Kb  = (bf16_t*)(ws + 24 * MB);  //  8 MB  K bf16 [S,D]
  bf16_t* Vt  = (bf16_t*)(ws + 32 * MB);  //  8 MB  V^T bf16 [D,S]
  bf16_t* AO  = (bf16_t*)(ws + 40 * MB);  //  8 MB  attn out bf16 [S,D]

  const int nX = S_LEN * D_MODEL;
  const int nW = D_MODEL * D_MODEL;
  cvt_f32_bf16<<<nX / 256, 256, 0, stream>>>(x, xb, nX);
  cvt_transpose_bf16<<<nW / 256, 256, 0, stream>>>(Wq_w, wtq, D_MODEL, D_MODEL);
  cvt_transpose_bf16<<<nW / 256, 256, 0, stream>>>(Wk_w, wtk, D_MODEL, D_MODEL);
  cvt_transpose_bf16<<<nW / 256, 256, 0, stream>>>(Wv_w, wtv, D_MODEL, D_MODEL);
  cvt_transpose_bf16<<<nW / 256, 256, 0, stream>>>(Wo_w, wto, D_MODEL, D_MODEL);

  // 32 M-blocks x 16 N-groups = 512 blocks x 8 waves
  gemm_bf16_wmma<<<512, 256, 0, stream>>>(xb, wtq, Wq_b, Qb, S_LEN, D_MODEL, D_MODEL, 0);
  gemm_bf16_wmma<<<512, 256, 0, stream>>>(xb, wtk, Wk_b, Kb, S_LEN, D_MODEL, D_MODEL, 0);
  gemm_bf16_wmma<<<512, 256, 0, stream>>>(xb, wtv, Wv_b, Vt, S_LEN, D_MODEL, D_MODEL, 1);

  // 16 heads x 32 q-blocks = 512 blocks x 8 waves
  attn_fwd_wmma<<<512, 256, 0, stream>>>(Qb, Kb, Vt, AO);

  gemm_bf16_wmma<<<512, 256, 0, stream>>>(AO, wto, Wo_b, d_out, S_LEN, D_MODEL, D_MODEL, 2);
}
